// TACA_70214125355294
// MI455X (gfx1250) — matrix operand order repaired and checked by reading, not compiled, not run on hardware
//
#include <hip/hip_runtime.h>
#include <hip/hip_bf16.h>

// ---------------------------------------------------------------------------
// Dual cross-attention (B=4, T=2048, D=512, H=8, DK=64) via CDNA5 WMMA bf16,
// with TDM (tensor_load_to_lds) double-buffered GEMM staging when available.
// ---------------------------------------------------------------------------

typedef __attribute__((ext_vector_type(16))) __bf16 v16bf;
typedef __attribute__((ext_vector_type(8)))  __bf16 v8bf;
typedef __attribute__((ext_vector_type(8)))  float  v8f;
typedef __attribute__((ext_vector_type(4)))  unsigned int uint32x4;
typedef __attribute__((ext_vector_type(8)))  int          int32x8;
typedef __attribute__((ext_vector_type(4)))  int          int32x4;

#define B_SZ   4
#define T_SZ   2048
#define D_SZ   512
#define H_SZ   8
#define DK_SZ  64
#define M_TOT  (B_SZ * T_SZ)          // 8192
#define NEGV   (-1.0e30f)
#define QSCALE 0.1875f                // 1.5 / sqrt(64)

#ifndef __has_builtin
#define __has_builtin(x) 0
#endif
#if defined(__HIP_DEVICE_COMPILE__) && \
    __has_builtin(__builtin_amdgcn_tensor_load_to_lds) && \
    __has_builtin(__builtin_amdgcn_s_wait_tensorcnt)
#define USE_TDM 1
#else
#define USE_TDM 0
#endif

__device__ inline v8f vzero8() {
    v8f z;
#pragma unroll
    for (int i = 0; i < 8; ++i) z[i] = 0.0f;
    return z;
}

__device__ inline v16bf make16(const __bf16* lo, const __bf16* hi) {
    v8bf a = *(const v8bf*)lo;
    v8bf b = *(const v8bf*)hi;
    v16bf r;
#pragma unroll
    for (int i = 0; i < 8; ++i) { r[i] = a[i]; r[i + 8] = b[i]; }
    return r;
}

__device__ inline void wmma_bf16(v8f& acc, v16bf a, v16bf b) {
    acc = __builtin_amdgcn_wmma_f32_16x16x32_bf16(
        false, a, false, b, (short)0, acc, false, false);
}

#if USE_TDM
// Async-copy a 2D tile (rows x cols bf16, global row stride g_stride elements)
// into LDS at byte offset lds_off, padding each row by 4 DWORDs (16B) so the
// LDS row stride is cols+8 elements. Tracked by TENSORcnt.
__device__ inline void tdm_load_2d(unsigned int lds_off, const __bf16* gptr,
                                   int rows, int cols, int g_stride,
                                   int tensor_rows) {
    unsigned long long ga = (unsigned long long)(size_t)gptr;
    uint32x4 g0;
    g0[0] = 1u;                                   // count=1, user descriptor
    g0[1] = lds_off;                              // lds_addr
    g0[2] = (unsigned int)ga;                     // global_addr[31:0]
    g0[3] = (unsigned int)((ga >> 32) & 0x1FFFFFFu) | (2u << 30); // addr[56:32], type=2
    int32x8 g1;
    unsigned int padint = (cols == 32) ? 3u : 4u; // 16 / 32 DWORDs per row
    g1[0] = (int)((1u << 16)      |               // data_size = 2 bytes
                  (1u << 20)      |               // pad_enable
                  (padint << 22)  |               // pad_interval
                  (3u << 25));                    // pad_amount = 4 DWORDs (16B)
    g1[1] = (int)(((unsigned int)g_stride & 0xFFFFu) << 16);          // tensor_dim0[15:0]
    g1[2] = (int)((((unsigned int)g_stride >> 16) & 0xFFFFu) |
                  (((unsigned int)tensor_rows & 0xFFFFu) << 16));     // dim0 hi | dim1 lo
    g1[3] = (int)((((unsigned int)tensor_rows >> 16) & 0xFFFFu) |
                  (((unsigned int)cols & 0xFFFFu) << 16));            // dim1 hi | tile_dim0
    g1[4] = (int)((unsigned int)rows & 0xFFFFu);                      // tile_dim1
    g1[5] = (int)(unsigned int)g_stride;                              // dim0_stride[31:0]
    g1[6] = 0;
    g1[7] = 0;
    int32x4 z4;
    z4[0] = z4[1] = z4[2] = z4[3] = 0;
#if __clang_major__ >= 23
    int32x8 z8;
#pragma unroll
    for (int i = 0; i < 8; ++i) z8[i] = 0;
    __builtin_amdgcn_tensor_load_to_lds(g0, g1, z4, z4, z8, 0);
#else
    __builtin_amdgcn_tensor_load_to_lds(g0, g1, z4, z4, 0);
#endif
}
#endif

// ---------------------------------------------------------------------------
// fp32 -> bf16 conversion (activations)
// ---------------------------------------------------------------------------
__global__ void conv_to_bf16(const float* __restrict__ src,
                             __bf16* __restrict__ dst, int n) {
    int i = (blockIdx.x * 256 + threadIdx.x) * 4;
    if (i + 3 < n) {
        float4 f = *(const float4*)(src + i);
        dst[i + 0] = (__bf16)f.x;
        dst[i + 1] = (__bf16)f.y;
        dst[i + 2] = (__bf16)f.z;
        dst[i + 3] = (__bf16)f.w;
    }
}

// ---------------------------------------------------------------------------
// fp32 [512,512] -> bf16 transposed [512,512]: dst[k*512+n] = src[n*512+k]
// 32x32 LDS tile transpose, block 256 = 32x8, grid (16,16).
// ---------------------------------------------------------------------------
__global__ __launch_bounds__(256)
void convT_to_bf16(const float* __restrict__ src, __bf16* __restrict__ dst) {
    __shared__ float tile[32][33];
    const int bx = blockIdx.x * 32, by = blockIdx.y * 32;
    const int tx = threadIdx.x & 31, ty = threadIdx.x >> 5;
#pragma unroll
    for (int i = 0; i < 32; i += 8)
        tile[ty + i][tx] = src[(size_t)(by + ty + i) * D_SZ + bx + tx];
    __syncthreads();
#pragma unroll
    for (int i = 0; i < 32; i += 8)
        dst[(size_t)(bx + ty + i) * D_SZ + by + tx] = (__bf16)tile[tx][ty + i];
}

// ---------------------------------------------------------------------------
// GEMM: out[m,n] = sum_k A[m,k] * Wt[k,n]  (A: [8192,512] bf16, Wt: [512,512] bf16,
// Wt already transposed). Block 128 (4 waves), 64x64 tile, K-step 64,
// TDM double-buffered staging (fallback: 16B vector copies).
//   mode 0: Q (scaled) -> [BH,T,64]   mode 1: K^T -> [BH,64,T]
//   mode 2: V -> [BH,T,64]            mode 3: fp32 -> outf[m*512+n]
// ---------------------------------------------------------------------------
__global__ __launch_bounds__(128)
void gemm_wmma(const __bf16* __restrict__ A, const __bf16* __restrict__ Wt,
               __bf16* __restrict__ outb, float* __restrict__ outf,
               int mode, float scale) {
    __shared__ __attribute__((aligned(16))) __bf16 lds_a[2][64 * 72]; // pad stride 72
    __shared__ __attribute__((aligned(16))) __bf16 lds_b[2][64 * 72];

    const int tid  = threadIdx.x;
    const int lane = tid & 31;
    const int wave = tid >> 5;
    const int half = lane >> 4;
    const int l16  = lane & 15;
    const int m0   = blockIdx.x * 64;
    const int n0   = blockIdx.y * 64;

    v8f acc[4];
#pragma unroll
    for (int c = 0; c < 4; ++c) acc[c] = vzero8();

#if USE_TDM
    if (wave == 0) {
        tdm_load_2d((unsigned int)(size_t)&lds_a[0][0], A + (size_t)m0 * D_SZ,
                    64, 64, D_SZ, M_TOT);
        tdm_load_2d((unsigned int)(size_t)&lds_b[0][0], Wt + n0,
                    64, 64, D_SZ, D_SZ);
    }
    for (int it = 0; it < 8; ++it) {
        const int cur = it & 1;
        if (wave == 0) {
            if (it < 7) {
                const int k0 = (it + 1) * 64;
                tdm_load_2d((unsigned int)(size_t)&lds_a[cur ^ 1][0],
                            A + (size_t)m0 * D_SZ + k0, 64, 64, D_SZ, M_TOT);
                tdm_load_2d((unsigned int)(size_t)&lds_b[cur ^ 1][0],
                            Wt + (size_t)k0 * D_SZ + n0, 64, 64, D_SZ, D_SZ);
                __builtin_amdgcn_s_wait_tensorcnt(2); // current tile landed
            } else {
                __builtin_amdgcn_s_wait_tensorcnt(0);
            }
        }
        __syncthreads();
        const __bf16* arow = &lds_a[cur][(wave * 16 + l16) * 72];
#pragma unroll
        for (int ks = 0; ks < 2; ++ks) {
            v16bf av = make16(arow + ks * 32 + 8 * half,
                              arow + ks * 32 + 16 + 8 * half);
#pragma unroll
            for (int c = 0; c < 4; ++c) {
                const __bf16* brow = &lds_b[cur][(ks * 32 + 8 * half) * 72 + c * 16 + l16];
                v16bf bv;
#pragma unroll
                for (int i = 0; i < 8; ++i) { bv[i] = brow[i * 72]; bv[i + 8] = brow[(16 + i) * 72]; }
                wmma_bf16(acc[c], av, bv);
            }
        }
        __syncthreads();  // reads done before next TDM overwrites this buffer
    }
#else
    for (int k0 = 0; k0 < D_SZ; k0 += 64) {
        if (k0 + 64 < D_SZ) {
            __builtin_prefetch(&A[(size_t)(m0 + (tid & 63)) * D_SZ + k0 + 64], 0, 3);
            __builtin_prefetch(&Wt[(size_t)(k0 + 64 + (tid & 63)) * D_SZ + n0], 0, 3);
        }
        // 16B vector copies: 64 rows x 8 chunks per tile, 4 chunks / thread.
#pragma unroll
        for (int i = 0; i < 4; ++i) {
            int idx = i * 128 + tid;          // 0..511
            int row = idx >> 3, ch = idx & 7;
            *(v8bf*)&lds_a[0][row * 72 + ch * 8] =
                *(const v8bf*)(A + (size_t)(m0 + row) * D_SZ + k0 + ch * 8);
            *(v8bf*)&lds_b[0][row * 72 + ch * 8] =
                *(const v8bf*)(Wt + (size_t)(k0 + row) * D_SZ + n0 + ch * 8);
        }
        __syncthreads();
        const __bf16* arow = &lds_a[0][(wave * 16 + l16) * 72];
#pragma unroll
        for (int ks = 0; ks < 2; ++ks) {
            v16bf av = make16(arow + ks * 32 + 8 * half,
                              arow + ks * 32 + 16 + 8 * half);
#pragma unroll
            for (int c = 0; c < 4; ++c) {
                const __bf16* brow = &lds_b[0][(ks * 32 + 8 * half) * 72 + c * 16 + l16];
                v16bf bv;
#pragma unroll
                for (int i = 0; i < 8; ++i) { bv[i] = brow[i * 72]; bv[i + 8] = brow[(16 + i) * 72]; }
                wmma_bf16(acc[c], av, bv);
            }
        }
        __syncthreads();
    }
#endif

    // Epilogue: D layout -> row M = r + 8*half (VGPR r), col N = c*16 + l16
    const int mbase = m0 + wave * 16 + 8 * half;
#pragma unroll
    for (int c = 0; c < 4; ++c) {
        int n = n0 + c * 16 + l16;
#pragma unroll
        for (int r = 0; r < 8; ++r) {
            int M = mbase + r;
            float val = acc[c][r] * scale;
            if (mode == 3) {
                outf[(size_t)M * D_SZ + n] = val;
            } else {
                int b = M >> 11, t = M & (T_SZ - 1);
                int h = n >> 6, dk = n & 63;
                if (mode == 1)
                    outb[((size_t)(b * H_SZ + h) * DK_SZ + dk) * T_SZ + t] = (__bf16)val;
                else
                    outb[((size_t)(b * H_SZ + h) * T_SZ + t) * DK_SZ + dk] = (__bf16)val;
            }
        }
    }
}

// ---------------------------------------------------------------------------
// Flash attention: grid(T/64, B*H), block 128 (4 waves x 16 q-rows).
// q: [BH,T,64] bf16 (pre-scaled), kt: [BH,64,T] bf16, v: [BH,T,64] bf16,
// mask: [B,T] int32, ctx: [B,T,512] bf16.  K/V stream from (L2-resident) global.
// ---------------------------------------------------------------------------
__global__ __launch_bounds__(128)
void attn_wmma(const __bf16* __restrict__ q, const __bf16* __restrict__ kt,
               const __bf16* __restrict__ v, const int* __restrict__ mask,
               __bf16* __restrict__ ctx) {
    __shared__ __attribute__((aligned(16))) __bf16 p_lds[4][16 * 72];

    const int tid  = threadIdx.x;
    const int lane = tid & 31;
    const int wave = tid >> 5;
    const int half = lane >> 4;
    const int l16  = lane & 15;
    const int bh   = blockIdx.y;             // b*8 + h
    const int b    = bh >> 3;
    const int h    = bh & 7;
    const int q0   = blockIdx.x * 64 + wave * 16;

    const __bf16* qrow = q + ((size_t)bh * T_SZ + q0 + l16) * DK_SZ;
    v16bf qa[2];
#pragma unroll
    for (int ds = 0; ds < 2; ++ds)
        qa[ds] = make16(qrow + ds * 32 + 8 * half, qrow + ds * 32 + 16 + 8 * half);

    float m_run[8], l_run[8];
    v8f acc_o[4];
#pragma unroll
    for (int r = 0; r < 8; ++r) { m_run[r] = -3.0e38f; l_run[r] = 0.0f; }
#pragma unroll
    for (int c = 0; c < 4; ++c) acc_o[c] = vzero8();

    const __bf16* ktb  = kt + (size_t)bh * DK_SZ * T_SZ;
    const __bf16* vb   = v  + (size_t)bh * T_SZ * DK_SZ;
    const int*    mrow = mask + (size_t)b * T_SZ;

    for (int kt0 = 0; kt0 < T_SZ; kt0 += 64) {
        v8f s[4];
#pragma unroll
        for (int c = 0; c < 4; ++c) s[c] = vzero8();
#pragma unroll
        for (int ds = 0; ds < 2; ++ds) {
            const __bf16* krow = ktb + (size_t)(ds * 32 + 8 * half) * T_SZ + kt0 + l16;
#pragma unroll
            for (int c = 0; c < 4; ++c) {
                v16bf bv;
#pragma unroll
                for (int i = 0; i < 8; ++i) { bv[i] = krow[(size_t)i * T_SZ + c * 16]; bv[i + 8] = krow[(size_t)(16 + i) * T_SZ + c * 16]; }
                wmma_bf16(s[c], qa[ds], bv);
            }
        }
#pragma unroll
        for (int c = 0; c < 4; ++c) {
            if (mrow[kt0 + c * 16 + l16] == 0) {
#pragma unroll
                for (int r = 0; r < 8; ++r) s[c][r] = NEGV;
            }
        }
#pragma unroll
        for (int r = 0; r < 8; ++r) {
            float mx = fmaxf(fmaxf(s[0][r], s[1][r]), fmaxf(s[2][r], s[3][r]));
#pragma unroll
            for (int off = 1; off < 16; off <<= 1)
                mx = fmaxf(mx, __shfl_xor(mx, off, 32));
            float mnew  = fmaxf(m_run[r], mx);
            float alpha = __expf(m_run[r] - mnew);
            float rs = 0.0f;
#pragma unroll
            for (int c = 0; c < 4; ++c) {
                float p = __expf(s[c][r] - mnew);
                s[c][r] = p;
                rs += p;
            }
#pragma unroll
            for (int off = 1; off < 16; off <<= 1)
                rs += __shfl_xor(rs, off, 32);
            m_run[r] = mnew;
            l_run[r] = l_run[r] * alpha + rs;
#pragma unroll
            for (int c = 0; c < 4; ++c) acc_o[c][r] = acc_o[c][r] * alpha;
        }
        __bf16* pw = &p_lds[wave][0];
#pragma unroll
        for (int c = 0; c < 4; ++c)
#pragma unroll
            for (int r = 0; r < 8; ++r)
                pw[(r + 8 * half) * 72 + c * 16 + l16] = (__bf16)s[c][r];
        __syncthreads();
#pragma unroll
        for (int kc = 0; kc < 2; ++kc) {
            const __bf16* prow = pw + l16 * 72 + kc * 32;
            v16bf pa = make16(prow + 8 * half, prow + 16 + 8 * half);
            const __bf16* vrow = vb + (size_t)(kt0 + kc * 32 + 8 * half) * DK_SZ + l16;
#pragma unroll
            for (int c = 0; c < 4; ++c) {
                v16bf bv;
#pragma unroll
                for (int i = 0; i < 8; ++i) { bv[i] = vrow[i * DK_SZ + c * 16]; bv[i + 8] = vrow[(16 + i) * DK_SZ + c * 16]; }
                wmma_bf16(acc_o[c], pa, bv);
            }
        }
        __syncthreads();
    }

#pragma unroll
    for (int r = 0; r < 8; ++r) {
        int t = q0 + r + 8 * half;
        float inv = 1.0f / l_run[r];
        size_t base = ((size_t)b * T_SZ + t) * D_SZ + h * DK_SZ;
#pragma unroll
        for (int c = 0; c < 4; ++c)
            ctx[base + c * 16 + l16] = (__bf16)(acc_o[c][r] * inv);
    }
}

// ---------------------------------------------------------------------------
// Host launcher
// ---------------------------------------------------------------------------
extern "C" void kernel_launch(void* const* d_in, const int* in_sizes, int n_in,
                              void* d_out, int out_size, void* d_ws, size_t ws_size,
                              hipStream_t stream) {
    (void)in_sizes; (void)n_in; (void)out_size; (void)ws_size;

    const float* skel      = (const float*)d_in[0];
    const float* sens      = (const float*)d_in[1];
    const int*   mask_skel = (const int*)d_in[2];
    const int*   mask_sens = (const int*)d_in[3];
    float*       out       = (float*)d_out;

    const size_t NBT = (size_t)B_SZ * T_SZ * D_SZ;   // 4,194,304 elements
    const size_t NW  = (size_t)D_SZ * D_SZ;          // 262,144 elements

    char* ws = (char*)d_ws;
    __bf16* xs = (__bf16*)ws;             ws += NBT * 2;   // skel bf16
    __bf16* xe = (__bf16*)ws;             ws += NBT * 2;   // sens bf16
    __bf16* wtb[8];                                        // transposed bf16 weights
    for (int i = 0; i < 8; ++i) { wtb[i] = (__bf16*)ws; ws += NW * 2; }
    __bf16* qb   = (__bf16*)ws;           ws += NBT * 2;   // Q   [BH,T,64]
    __bf16* ktb  = (__bf16*)ws;           ws += NBT * 2;   // K^T [BH,64,T]
    __bf16* vbuf = (__bf16*)ws;           ws += NBT * 2;   // V   [BH,T,64]
    __bf16* cxb  = (__bf16*)ws;           ws += NBT * 2;   // ctx [B,T,512]

    conv_to_bf16<<<dim3(NBT / 1024), dim3(256), 0, stream>>>(skel, xs, (int)NBT);
    conv_to_bf16<<<dim3(NBT / 1024), dim3(256), 0, stream>>>(sens, xe, (int)NBT);
    for (int i = 0; i < 8; ++i)
        convT_to_bf16<<<dim3(16, 16), dim3(256), 0, stream>>>(
            (const float*)d_in[4 + i], wtb[i]);

    const dim3 gg(M_TOT / 64, D_SZ / 64), gb(128);
    const dim3 ga(T_SZ / 64, B_SZ * H_SZ);

    // --- pass 0: s2e (skel queries over sensor K/V) -> tuple slot 1 ---
    gemm_wmma<<<gg, gb, 0, stream>>>(xs, wtb[0], qb,   nullptr, 0, QSCALE); // Wq_s2e
    gemm_wmma<<<gg, gb, 0, stream>>>(xe, wtb[1], ktb,  nullptr, 1, 1.0f);   // Wk_e
    gemm_wmma<<<gg, gb, 0, stream>>>(xe, wtb[2], vbuf, nullptr, 2, 1.0f);   // Wv_e
    attn_wmma<<<ga, gb, 0, stream>>>(qb, ktb, vbuf, mask_sens, cxb);
    gemm_wmma<<<gg, gb, 0, stream>>>(cxb, wtb[7], nullptr, out + NBT, 3, 1.0f); // Wo_e

    // --- pass 1: e2s (sensor queries over skeleton K/V) -> tuple slot 0 ---
    gemm_wmma<<<gg, gb, 0, stream>>>(xe, wtb[3], qb,   nullptr, 0, QSCALE); // Wq_e2s
    gemm_wmma<<<gg, gb, 0, stream>>>(xs, wtb[4], ktb,  nullptr, 1, 1.0f);   // Wk_s
    gemm_wmma<<<gg, gb, 0, stream>>>(xs, wtb[5], vbuf, nullptr, 2, 1.0f);   // Wv_s
    attn_wmma<<<ga, gb, 0, stream>>>(qb, ktb, vbuf, mask_skel, cxb);
    gemm_wmma<<<gg, gb, 0, stream>>>(cxb, wtb[6], nullptr, out, 3, 1.0f);   // Wo_s
}
